// DenseSOFLayer_32409823216060
// MI455X (gfx1250) — compile-verified, matrix operand order checked by reading, not run on hardware
//
#include <hip/hip_runtime.h>
#include <hip/hip_bf16.h>
#include <stdint.h>

// ---------------------------------------------------------------------------
// out[b,f] = -0.5 * sum_d ((x[b,d]-mu[f,d])/scale[f,d])^2
// GEMM form with K=2D: A=[x^2 | x] (B x 2048, f16), W=[-0.5*inv2 | mu*inv2]
// (F x 2048, f16), bias[f] = -0.5*sum(mu^2*inv2);  out = A @ W^T + bias (f32)
// ---------------------------------------------------------------------------

#define BATCH 8192
#define FEAT  4096
#define DDIM  1024
#define K2    2048   // 2*DDIM

#define BM 128
#define BN 128
#define BK 32
#define NKSTEP (K2 / BK)      // 64

// LDS tile: 128 rows x 32 f16 (64B = 16 dwords) + 4-dword TDM pad -> 80B pitch
#define LDS_PITCH 80
#define LDS_TILE  (128 * LDS_PITCH)   // 10240 bytes

typedef _Float16 v16h __attribute__((ext_vector_type(16)));
typedef _Float16 v8h  __attribute__((ext_vector_type(8)));
typedef float    v8f  __attribute__((ext_vector_type(8)));
typedef unsigned int v4u __attribute__((ext_vector_type(4)));
typedef int      v8i  __attribute__((ext_vector_type(8)));
typedef int      v4i  __attribute__((ext_vector_type(4)));

// ------------------------- prep kernels ------------------------------------

__global__ __launch_bounds__(256) void prep_x_kernel(const float* __restrict__ x,
                                                     _Float16* __restrict__ A) {
    int i = blockIdx.x * 256 + threadIdx.x;          // i over BATCH*DDIM
    int b = i >> 10;                                  // / DDIM
    int d = i & (DDIM - 1);
    float v = x[i];
    A[(size_t)b * K2 + d]        = (_Float16)(v * v);
    A[(size_t)b * K2 + DDIM + d] = (_Float16)v;
}

__global__ __launch_bounds__(256) void prep_w_kernel(const float* __restrict__ mu,
                                                     const float* __restrict__ sd,
                                                     _Float16* __restrict__ W) {
    int i = blockIdx.x * 256 + threadIdx.x;          // i over FEAT*DDIM
    int f = i >> 10;
    int d = i & (DDIM - 1);
    float s = sd[i];
    float inv2 = 1.0f / (s * s);
    float m = mu[i];
    W[(size_t)f * K2 + d]        = (_Float16)(-0.5f * inv2);
    W[(size_t)f * K2 + DDIM + d] = (_Float16)(m * inv2);
}

__global__ __launch_bounds__(256) void prep_bias_kernel(const float* __restrict__ mu,
                                                        const float* __restrict__ sd,
                                                        float* __restrict__ bias) {
    __shared__ float red[256];
    int f = blockIdx.x;
    int t = threadIdx.x;
    float acc = 0.0f;
    for (int d = t; d < DDIM; d += 256) {
        float s = sd[(size_t)f * DDIM + d];
        float m = mu[(size_t)f * DDIM + d];
        acc += (m * m) / (s * s);
    }
    red[t] = acc;
    __syncthreads();
    for (int off = 128; off > 0; off >>= 1) {
        if (t < off) red[t] += red[t + off];
        __syncthreads();
    }
    if (t == 0) bias[f] = -0.5f * red[0];
}

// ------------------------- TDM descriptor ----------------------------------
// 2D tensor (rows x K2 f16, row-major). Tile = 32 cols x 128 rows starting at
// gaddr. Pad LDS dest every 16 dwords (one 32-f16 row) by 4 dwords -> 80B pitch.

__device__ __forceinline__ void tdm_load_tile(unsigned long long gaddr,
                                              unsigned lds_addr,
                                              unsigned tensor_rows) {
    v4u g0;
    g0.x = 1u;                                            // count=1 (valid)
    g0.y = lds_addr;                                      // LDS byte address
    g0.z = (unsigned)(gaddr & 0xffffffffull);             // global_addr[31:0]
    g0.w = (unsigned)((gaddr >> 32) & 0x01ffffffull)      // global_addr[56:32]
         | (2u << 30);                                    // type = 2 (image)

    unsigned d0 = (1u << 16)      // data_size = 1 -> 2 bytes
                | (1u << 20)      // pad_enable
                | (3u << 22)      // pad_interval code 3 -> 16 dwords
                | (3u << 25);     // pad_amount   code 3 -> 4 dwords
    unsigned d1 = (unsigned)(K2 & 0xffff) << 16;          // tensor_dim0[15:0]
    unsigned d2 = (unsigned)(K2 >> 16)
                | ((tensor_rows & 0xffffu) << 16);        // tensor_dim1[15:0]
    unsigned d3 = (tensor_rows >> 16)
                | ((unsigned)BK << 16);                   // tile_dim0 = 32
    unsigned d4 = 128u;                                   // tile_dim1=128, tile_dim2=0
    unsigned d5 = (unsigned)K2;                           // tensor_dim0_stride
    v8i g1 = { (int)d0, (int)d1, (int)d2, (int)d3,
               (int)d4, (int)d5, 0, 0 };
    v4i gz4 = { 0, 0, 0, 0 };
    v8i gz8 = { 0, 0, 0, 0, 0, 0, 0, 0 };
    __builtin_amdgcn_tensor_load_to_lds(g0, g1, gz4, gz4, gz8, 0);
}

// ------------------------- GEMM kernel -------------------------------------

__global__ __launch_bounds__(256) void sof_gemm_kernel(const _Float16* __restrict__ A,
                                                       const _Float16* __restrict__ W,
                                                       const float* __restrict__ bias,
                                                       float* __restrict__ out) {
    __shared__ __align__(16) unsigned char smem[4 * LDS_TILE];

    const int bn = blockIdx.x;          // feature-tile index  (FEAT/128)
    const int bm = blockIdx.y;          // batch-tile index    (BATCH/128)
    const int tid = threadIdx.x;
    const int lane = tid & 31;
    const int wave = tid >> 5;          // 8 waves
    const int wm = wave & 1;            // 2 wave-rows of 64
    const int wn = wave >> 1;           // 4 wave-cols of 32
    const int lh = lane >> 4;           // half-wave select
    const int lm = lane & 15;

    const unsigned ldsBase = (unsigned)(uintptr_t)(void*)&smem[0];
    const unsigned bufA[2] = { ldsBase,                ldsBase + LDS_TILE };
    const unsigned bufW[2] = { ldsBase + 2 * LDS_TILE, ldsBase + 3 * LDS_TILE };

    const unsigned long long aBase = (unsigned long long)(uintptr_t)A
                                   + (unsigned long long)bm * BM * K2 * 2ull;
    const unsigned long long wBase = (unsigned long long)(uintptr_t)W
                                   + (unsigned long long)bn * BN * K2 * 2ull;

    v8f acc[4][2] = {};

    // Prologue: stage first K-slab via TDM (wave 0 only; TENSORcnt is per-wave)
    if (tid < 32) {
        tdm_load_tile(aBase, bufA[0], BATCH);
        tdm_load_tile(wBase, bufW[0], FEAT);
        __builtin_amdgcn_s_wait_tensorcnt(0);
    }
    __syncthreads();

    for (int kt = 0; kt < NKSTEP; ++kt) {
        const int cur = kt & 1;

        // Issue next slab while computing the current one
        if ((kt + 1 < NKSTEP) && (tid < 32)) {
            unsigned long long koff = (unsigned long long)(kt + 1) * BK * 2ull;
            tdm_load_tile(aBase + koff, bufA[1 - cur], BATCH);
            tdm_load_tile(wBase + koff, bufW[1 - cur], FEAT);
        }

        const unsigned char* As = smem + (bufA[cur] - ldsBase);
        const unsigned char* Ws = smem + (bufW[cur] - ldsBase);

        // A fragments: lane<16 -> K {0..7,16..23}; lane>=16 -> K {8..15,24..31}
        v16h afrag[4], bfrag[2];
        union V16 { v16h v; v8h h[2]; };
#pragma unroll
        for (int mi = 0; mi < 4; ++mi) {
            int row = wm * 64 + mi * 16 + lm;
            const unsigned char* rp = As + row * LDS_PITCH;
            V16 t;
            t.h[0] = *(const v8h*)(rp + lh * 16);
            t.h[1] = *(const v8h*)(rp + 32 + lh * 16);
            afrag[mi] = t.v;
        }
#pragma unroll
        for (int ni = 0; ni < 2; ++ni) {
            int row = wn * 32 + ni * 16 + lm;       // row of W == column of B
            const unsigned char* rp = Ws + row * LDS_PITCH;
            V16 t;
            t.h[0] = *(const v8h*)(rp + lh * 16);
            t.h[1] = *(const v8h*)(rp + 32 + lh * 16);
            bfrag[ni] = t.v;
        }

#pragma unroll
        for (int mi = 0; mi < 4; ++mi)
#pragma unroll
            for (int ni = 0; ni < 2; ++ni)
                acc[mi][ni] = __builtin_amdgcn_wmma_f32_16x16x32_f16(
                    /*neg_a=*/false, afrag[mi],
                    /*neg_b=*/false, bfrag[ni],
                    /*c_mod=*/(short)0, acc[mi][ni],
                    /*reuse_a=*/false, /*reuse_b=*/false);

        if ((kt + 1 < NKSTEP) && (tid < 32))
            __builtin_amdgcn_s_wait_tensorcnt(0);
        __syncthreads();
    }

    // Epilogue: C/D layout -> M = vgpr + 8*lh, N = lm; fuse bias.
    float bv[2];
#pragma unroll
    for (int ni = 0; ni < 2; ++ni)
        bv[ni] = bias[bn * BN + wn * 32 + ni * 16 + lm];

#pragma unroll
    for (int mi = 0; mi < 4; ++mi) {
        int row0 = bm * BM + wm * 64 + mi * 16 + lh * 8;
#pragma unroll
        for (int ni = 0; ni < 2; ++ni) {
            int col = bn * BN + wn * 32 + ni * 16 + lm;
            float* op = out + (size_t)row0 * FEAT + col;
#pragma unroll
            for (int r = 0; r < 8; ++r)
                op[(size_t)r * FEAT] = acc[mi][ni][r] + bv[ni];
        }
    }
}

// ------------------------- launch ------------------------------------------

extern "C" void kernel_launch(void* const* d_in, const int* in_sizes, int n_in,
                              void* d_out, int out_size, void* d_ws, size_t ws_size,
                              hipStream_t stream) {
    const float* x  = (const float*)d_in[0];   // [8192,1024]
    const float* mu = (const float*)d_in[1];   // [4096,1024]
    const float* sd = (const float*)d_in[2];   // [4096,1024]
    float* out = (float*)d_out;                // [8192,4096]

    char* ws = (char*)d_ws;
    _Float16* A = (_Float16*)ws;                                        // 32 MiB
    _Float16* W = (_Float16*)(ws + (size_t)BATCH * K2 * 2);             // 16 MiB
    float* bias = (float*)(ws + (size_t)BATCH * K2 * 2
                              + (size_t)FEAT * K2 * 2);                 // 16 KiB

    prep_x_kernel<<<(BATCH * DDIM) / 256, 256, 0, stream>>>(x, A);
    prep_w_kernel<<<(FEAT * DDIM) / 256, 256, 0, stream>>>(mu, sd, W);
    prep_bias_kernel<<<FEAT, 256, 0, stream>>>(mu, sd, bias);

    dim3 grid(FEAT / BN, BATCH / BM);   // (32, 64)
    sof_gemm_kernel<<<grid, 256, 0, stream>>>(A, W, bias, out);
}
